// GCN_3_layer_edge_weight_fc_45311904973172
// MI455X (gfx1250) — compile-verified, matrix operand order checked by reading, not run on hardware
//
#include <hip/hip_runtime.h>
#include <hip/hip_bf16.h>
#include <stdint.h>

// ---------- CDNA5 WMMA types ----------
typedef __attribute__((ext_vector_type(16))) __bf16 v16bf;
typedef __attribute__((ext_vector_type(8)))  __bf16 v8bf;
typedef __attribute__((ext_vector_type(8)))  float  v8f;

static __device__ __forceinline__ __bf16 f2bf(float x) { return (__bf16)x; }

// ============================================================
// Zero-fill (float4 granularity)
// ============================================================
__global__ __launch_bounds__(256) void zero_f4(float4* __restrict__ p, int n4) {
    int i = blockIdx.x * 256 + threadIdx.x;
    if (i < n4) p[i] = make_float4(0.f, 0.f, 0.f, 0.f);
}

// ============================================================
// One-time weight prep: W[k][n] (f32, K=128 x Nw) ->
// Wt[n][k] (bf16, NPAD rows x 128, zero-padded n >= Nw)
// ============================================================
__global__ __launch_bounds__(256) void prep_wt(
    const float* __restrict__ W, __bf16* __restrict__ out, int Nw, int total) {
    int idx = blockIdx.x * 256 + threadIdx.x;
    if (idx >= total) return;
    int n = idx >> 7;
    int k = idx & 127;
    float v = (n < Nw) ? W[(size_t)k * Nw + n] : 0.f;
    out[idx] = f2bf(v);
}

// ============================================================
// Edge scatter: agg[dst[e]] += x[src[e]] * ew[e]
// one wave32 per edge, one float4 (4 features) per lane -> 128 features
// ============================================================
__global__ __launch_bounds__(256) void scatter_edges(
    const float* __restrict__ x, const long long* __restrict__ src,
    const long long* __restrict__ dst, const float* __restrict__ ew,
    float* __restrict__ agg, int n_edges) {
    int gid  = blockIdx.x * 256 + threadIdx.x;
    int e    = gid >> 5;
    int lane = gid & 31;
    if (e >= n_edges) return;
    long long s = src[e];
    long long d = dst[e];
    float     w = ew[e];
    const float4* xr = (const float4*)(x + (size_t)s * 128);
    float4 v = xr[lane];
    float* out = agg + (size_t)d * 128 + lane * 4;
    atomicAdd(out + 0, v.x * w);
    atomicAdd(out + 1, v.y * w);
    atomicAdd(out + 2, v.z * w);
    atomicAdd(out + 3, v.w * w);
}

// ============================================================
// K=128 GEMM via v_wmma_f32_16x16x32_bf16.
//   C[M,Nw] = act( A[M,128] @ W[128,Nw] + bias )
// Wt: pre-transposed bf16 weights, NPAD rows x 128 cols, contiguous.
// Staged into LDS (row stride 136 elems for banking) with CDNA5
// global_load_async_to_lds_b128 (ASYNCcnt), then consumed as B frags
// via ds_load_b128. Each wave computes TPW consecutive M-tiles to
// amortize staging. Block = 256 threads = 8 wave32.
// ============================================================
template <int NT, bool RELU_OUT, bool FUSE_ADD_IN, int TPW>
__global__ __launch_bounds__(256) void gemm_k128_bf16(
    const float* __restrict__ A0, const float* __restrict__ A1,
    const __bf16* __restrict__ Wt, const float* __restrict__ bias,
    float* __restrict__ C, int M, int Nw) {

    constexpr int NPAD = NT * 16;
    __shared__ alignas(16) __bf16 Wl[NPAD][136];   // 272 B row stride
    __shared__ float bs[NPAD];

    const int tid = threadIdx.x;
    const unsigned ldsbase = (unsigned)(size_t)(&Wl[0][0]);

    // ---- async stage Wt (NPAD x 256 B) into LDS, 16 B per lane ----
    constexpr int CHUNKS = NPAD * 16;              // 16-byte chunks
#pragma unroll
    for (int c = tid; c < CHUNKS; c += 256) {
        int row = c >> 4;
        int col = (c & 15) * 16;
        unsigned loff = ldsbase + (unsigned)(row * 272 + col);
        unsigned goff = (unsigned)c * 16u;
        asm volatile("global_load_async_to_lds_b128 %0, %1, %2 offset:0"
                     :: "v"(loff), "v"(goff), "s"(Wt) : "memory");
    }
    if (tid < NPAD) bs[tid] = (tid < Nw) ? bias[tid] : 0.f;
    asm volatile("s_wait_asynccnt 0" ::: "memory");
    __syncthreads();

    const int wave = tid >> 5;
    const int lane = tid & 31;
    const int g  = lane >> 4;                      // half-wave group
    const int mr = lane & 15;                      // A row within tile
    const int nl = lane & 15;                      // B/C column within tile
    const int mtiles = M >> 4;                     // M multiple of 16
    const int mt0 = blockIdx.x * (8 * TPW) + wave * TPW;

    for (int t = 0; t < TPW; ++t) {
        const int mt = mt0 + t;
        if (mt >= mtiles) break;                   // wave-uniform exit

        v8f acc[NT];
#pragma unroll
        for (int nt = 0; nt < NT; ++nt) {
            float bv = bs[nt * 16 + nl];
#pragma unroll
            for (int v = 0; v < 8; ++v) acc[nt][v] = bv;
        }

        const float* ar0 = A0 + (size_t)(mt * 16 + mr) * 128;
        const float* ar1 = FUSE_ADD_IN ? (A1 + (size_t)(mt * 16 + mr) * 128) : nullptr;

#pragma unroll
        for (int kt = 0; kt < 4; ++kt) {
            const int k0 = kt * 32 + g * 8;        // lane's first K run
            // A fragment: k in [k0, k0+8) and [k0+16, k0+24)
            float4 p0 = *(const float4*)(ar0 + k0);
            float4 p1 = *(const float4*)(ar0 + k0 + 4);
            float4 q0 = *(const float4*)(ar0 + k0 + 16);
            float4 q1 = *(const float4*)(ar0 + k0 + 20);
            if (FUSE_ADD_IN) {
                float4 r0 = *(const float4*)(ar1 + k0);
                float4 r1 = *(const float4*)(ar1 + k0 + 4);
                float4 s0 = *(const float4*)(ar1 + k0 + 16);
                float4 s1 = *(const float4*)(ar1 + k0 + 20);
                p0.x = fmaxf(p0.x + r0.x, 0.f); p0.y = fmaxf(p0.y + r0.y, 0.f);
                p0.z = fmaxf(p0.z + r0.z, 0.f); p0.w = fmaxf(p0.w + r0.w, 0.f);
                p1.x = fmaxf(p1.x + r1.x, 0.f); p1.y = fmaxf(p1.y + r1.y, 0.f);
                p1.z = fmaxf(p1.z + r1.z, 0.f); p1.w = fmaxf(p1.w + r1.w, 0.f);
                q0.x = fmaxf(q0.x + s0.x, 0.f); q0.y = fmaxf(q0.y + s0.y, 0.f);
                q0.z = fmaxf(q0.z + s0.z, 0.f); q0.w = fmaxf(q0.w + s0.w, 0.f);
                q1.x = fmaxf(q1.x + s1.x, 0.f); q1.y = fmaxf(q1.y + s1.y, 0.f);
                q1.z = fmaxf(q1.z + s1.z, 0.f); q1.w = fmaxf(q1.w + s1.w, 0.f);
            }
            v16bf a;
            a[0]  = f2bf(p0.x); a[1]  = f2bf(p0.y); a[2]  = f2bf(p0.z); a[3]  = f2bf(p0.w);
            a[4]  = f2bf(p1.x); a[5]  = f2bf(p1.y); a[6]  = f2bf(p1.z); a[7]  = f2bf(p1.w);
            a[8]  = f2bf(q0.x); a[9]  = f2bf(q0.y); a[10] = f2bf(q0.z); a[11] = f2bf(q0.w);
            a[12] = f2bf(q1.x); a[13] = f2bf(q1.y); a[14] = f2bf(q1.z); a[15] = f2bf(q1.w);

#pragma unroll
            for (int nt = 0; nt < NT; ++nt) {
                v8bf b0 = *(const v8bf*)(&Wl[nt * 16 + nl][k0]);
                v8bf b1 = *(const v8bf*)(&Wl[nt * 16 + nl][k0 + 16]);
                v16bf b;
#pragma unroll
                for (int i = 0; i < 8; ++i) { b[i] = b0[i]; b[8 + i] = b1[i]; }
                acc[nt] = __builtin_amdgcn_wmma_f32_16x16x32_bf16(
                    false, a, false, b, (short)0, acc[nt], false, false);
            }
        }

        // ---- store: VGPR v -> row mt*16 + v + 8*g, col nt*16 + nl ----
#pragma unroll
        for (int nt = 0; nt < NT; ++nt) {
            int n = nt * 16 + nl;
            if (n < Nw) {
#pragma unroll
                for (int v = 0; v < 8; ++v) {
                    int row = mt * 16 + g * 8 + v;
                    float val = acc[nt][v];
                    if (RELU_OUT) val = fmaxf(val, 0.f);
                    C[(size_t)row * Nw + n] = val;
                }
            }
        }
    }
}

// ============================================================
// Host-side orchestration
// ============================================================
extern "C" void kernel_launch(void* const* d_in, const int* in_sizes, int n_in,
                              void* d_out, int out_size, void* d_ws, size_t ws_size,
                              hipStream_t stream) {
    const float*     x   = (const float*)d_in[0];
    const long long* src = (const long long*)d_in[1];
    const long long* dst = (const long long*)d_in[2];
    const float*     ew  = (const float*)d_in[3];
    const float *Wr = (const float*)d_in[4],  *br  = (const float*)d_in[5];
    const float *W1 = (const float*)d_in[6],  *b1  = (const float*)d_in[7];
    const float *W2 = (const float*)d_in[8],  *b2  = (const float*)d_in[9];
    const float *W3 = (const float*)d_in[10], *b3  = (const float*)d_in[11];
    const float *Wop= (const float*)d_in[12], *bop = (const float*)d_in[13];
    float* out = (float*)d_out;

    const int n_nodes = in_sizes[0] / 128;
    const int n_edges = in_sizes[1];
    const size_t feat = (size_t)n_nodes * 128;

    float* agg    = (float*)d_ws;
    float* bufH   = agg  + feat;
    float* bufRes = bufH + feat;
    __bf16* wtR = (__bf16*)(bufRes + feat);
    __bf16* wt1 = wtR + 128 * 128;
    __bf16* wt2 = wt1 + 128 * 128;
    __bf16* wt3 = wt2 + 128 * 128;
    __bf16* wtO = wt3 + 128 * 128;                 // 48 x 128

    const dim3 blk(256);

    // ---- one-time weight transpose + bf16 conversion ----
    prep_wt<<<64, blk, 0, stream>>>(Wr,  wtR, 128, 128 * 128);
    prep_wt<<<64, blk, 0, stream>>>(W1,  wt1, 128, 128 * 128);
    prep_wt<<<64, blk, 0, stream>>>(W2,  wt2, 128, 128 * 128);
    prep_wt<<<64, blk, 0, stream>>>(W3,  wt3, 128, 128 * 128);
    prep_wt<<<24, blk, 0, stream>>>(Wop, wtO, 40,  48 * 128);

    const int mtiles = n_nodes / 16;
    const dim3 gemmGrid((mtiles + 31) / 32);       // 8 waves x 4 tiles per block
    const int n4 = (int)(feat / 4);
    const dim3 zeroGrid((n4 + 255) / 256);
    const dim3 scatGrid(((size_t)n_edges * 32 + 255) / 256);

    // residual path: res = x @ Wr + br
    gemm_k128_bf16<8, false, false, 4><<<gemmGrid, blk, 0, stream>>>(
        x, nullptr, wtR, br, bufRes, n_nodes, 128);

    // layer 1: h = relu((scatter x) @ W1 + b1)
    zero_f4<<<zeroGrid, blk, 0, stream>>>((float4*)agg, n4);
    scatter_edges<<<scatGrid, blk, 0, stream>>>(x, src, dst, ew, agg, n_edges);
    gemm_k128_bf16<8, true, false, 4><<<gemmGrid, blk, 0, stream>>>(
        agg, nullptr, wt1, b1, bufH, n_nodes, 128);

    // layer 2
    zero_f4<<<zeroGrid, blk, 0, stream>>>((float4*)agg, n4);
    scatter_edges<<<scatGrid, blk, 0, stream>>>(bufH, src, dst, ew, agg, n_edges);
    gemm_k128_bf16<8, true, false, 4><<<gemmGrid, blk, 0, stream>>>(
        agg, nullptr, wt2, b2, bufH, n_nodes, 128);

    // layer 3 (no relu yet; fused with residual below)
    zero_f4<<<zeroGrid, blk, 0, stream>>>((float4*)agg, n4);
    scatter_edges<<<scatGrid, blk, 0, stream>>>(bufH, src, dst, ew, agg, n_edges);
    gemm_k128_bf16<8, false, false, 4><<<gemmGrid, blk, 0, stream>>>(
        agg, nullptr, wt3, b3, bufH, n_nodes, 128);

    // output: out = relu(h3 + res) @ Wop + bop   (N = 40, padded to 48)
    gemm_k128_bf16<3, false, true, 4><<<gemmGrid, blk, 0, stream>>>(
        bufH, bufRes, wtO, bop, out, n_nodes, 40);
}
